// GNN_BP4_41326175322289
// MI455X (gfx1250) — matrix operand order repaired and checked by reading, not compiled, not run on hardware
//
#include <hip/hip_runtime.h>

// ---------------------------------------------------------------------------
// GNN BP message passing for MI455X (gfx1250), wave32 + WMMA f16 path.
//
// Problem sizes (from reference):
//   B=4, NUM_VN=65536, NUM_CN=32768, D=32, E=262144, HID=40, MSG=20
//
// Pipeline per side:
//   zero m [B,NUM_CN,20]  (workspace)
//   edge kernel : gather feat[16x64] -> relu(feat@Wm1)[16x40] -> @Wm2 [16x20]
//                 -> atomicAdd scatter into m
//   node kernel : [m | h_to | logit] [16x53] -> relu(@We1)[16x40] -> @We2 [16x32]
//                 -> store to output
// All GEMMs via v_wmma_f32_16x16x32_f16 (f32 accumulate).
//
// Key optimization notes:
//  * All operand tensors are L2-resident (192 MB L2); kernel is latency /
//    atomic bound, so each wave keeps all weight B-fragments in registers and
//    grid-strides over 16-row tiles.
//  * ReLU write-back is unconditional for all 48 columns: columns 40..47 are
//    K-rows whose layer-2 weight fragments are zero-padded, so garbage there
//    is annihilated — this removes exec-mask branches from the hot loop.
// ---------------------------------------------------------------------------

typedef __attribute__((ext_vector_type(16))) _Float16 v16h;
typedef __attribute__((ext_vector_type(8)))  _Float16 v8h;
typedef __attribute__((ext_vector_type(8)))  float    v8f;

namespace {
constexpr int kB      = 4;
constexpr int kNumVN  = 65536;
constexpr int kNumCN  = 32768;
constexpr int kD      = 32;
constexpr int kE      = 262144;
constexpr int kHID    = 40;
constexpr int kMSG    = 20;

constexpr int kWaves     = 8;    // 256 threads / wave32
constexpr int kLdsStride = 80;   // halves per tile row (160B: 16B aligned, bank-spread)
}

#define WMMA_F16(A, Bf, C) \
  __builtin_amdgcn_wmma_f32_16x16x32_f16(false, (A), false, (Bf), (short)0, (C), false, false)

// Load a 32x16 f16 B-fragment from a row-major f32 weight matrix W[K x N], ld = N.
// Zero-pads K/N beyond (Kvalid, Nvalid).
__device__ __forceinline__ v16h load_bfrag(const float* __restrict__ W, int ld,
                                           int Kvalid, int Nvalid,
                                           int kblock, int ntile, int lane) {
  const int n      = ntile * 16 + (lane & 15);
  const int kstart = kblock * 32 + (lane >> 4) * 8;
  v16h r;
#pragma unroll
  for (int i = 0; i < 16; ++i) {
    const int k = kstart + (i < 8 ? i : i + 8);
    float w = 0.0f;
    if (k < Kvalid && n < Nvalid) w = W[k * ld + n];
    r[i] = (_Float16)w;
  }
  return r;
}

// Load a 16x32 f16 A-fragment (rows = tile rows, K = kblock*32..+31) from an LDS
// tile with row stride kLdsStride halves.
__device__ __forceinline__ v16h load_afrag(const _Float16* tile, int kblock, int lane) {
  const int row = lane & 15;
  const int k0  = kblock * 32 + (lane >> 4) * 8;
  const _Float16* p = tile + row * kLdsStride + k0;
  const v8h a0 = *(const v8h*)(p);        // K = k0 .. k0+7
  const v8h a1 = *(const v8h*)(p + 16);   // K = k0+16 .. k0+23
  v16h r;
#pragma unroll
  for (int i = 0; i < 8; ++i) { r[i] = a0[i]; r[8 + i] = a1[i]; }
  return r;
}

__global__ void bp_zero_kernel(float* __restrict__ p, int n) {
  const int i = blockIdx.x * blockDim.x + threadIdx.x;
  if (i < n) p[i] = 0.0f;
}

// ---------------------------------------------------------------------------
// Edge kernel: one wave processes a tile of 16 edges of one batch element.
// ---------------------------------------------------------------------------
__global__ __launch_bounds__(256) void bp_edge_kernel(
    const float* __restrict__ h_from,   // [B, NUM_VN, D]
    const float* __restrict__ h_to,     // [B, NUM_CN, D]
    const int*   __restrict__ from_ind, // [E]
    const int*   __restrict__ to_ind,   // [E]
    const float* __restrict__ Wm1,      // [2D=64, HID=40]
    const float* __restrict__ Wm2,      // [HID=40, MSG=20]
    float*       __restrict__ m)        // [B, NUM_CN, MSG]
{
  __shared__ __align__(16) _Float16 s_feat[kWaves][16 * kLdsStride];
  __shared__ __align__(16) _Float16 s_hid [kWaves][16 * kLdsStride];
  __shared__ int s_to[kWaves][16];

  const int wave = threadIdx.x >> 5;
  const int lane = threadIdx.x & 31;
  const int lw   = lane & 15;
  const int hi   = lane >> 4;

  _Float16* feat = s_feat[wave];
  _Float16* hid  = s_hid[wave];

  // Zero LDS tiles once (keeps K pad regions >= 48 zero forever).
  for (int i = lane; i < 16 * kLdsStride; i += 32) {
    feat[i] = (_Float16)0.0f;
    hid[i]  = (_Float16)0.0f;
  }

  // Per-wave resident weight fragments (reused across all tiles).
  v16h b1[2][3];  // Wm1: K=64 (2 kblocks), N=40 (3 ntiles)
#pragma unroll
  for (int kb = 0; kb < 2; ++kb)
#pragma unroll
    for (int nt = 0; nt < 3; ++nt)
      b1[kb][nt] = load_bfrag(Wm1, kHID, 2 * kD, kHID, kb, nt, lane);
  v16h b2[2][2];  // Wm2: K=40 pad 64 (zeros), N=20 pad 32
#pragma unroll
  for (int kb = 0; kb < 2; ++kb)
#pragma unroll
    for (int nt = 0; nt < 2; ++nt)
      b2[kb][nt] = load_bfrag(Wm2, kMSG, kHID, kMSG, kb, nt, lane);

  constexpr int kTilesPerB = kE / 16;            // 16384
  constexpr int kTiles     = kB * kTilesPerB;    // 65536
  const int wid = blockIdx.x * kWaves + wave;
  const int nw  = gridDim.x * kWaves;

  for (int t = wid; t < kTiles; t += nw) {
    const int b  = t / kTilesPerB;
    const int e0 = (t % kTilesPerB) * 16;
    const int e  = e0 + lw;

    // Gather: lanes 0-15 fetch h_from rows (cols 0..31),
    //         lanes 16-31 fetch h_to rows (cols 32..63) + cache to_ind.
    if (hi == 0) {
      const int src = from_ind[e];
      const float4* p = (const float4*)(h_from + ((size_t)b * kNumVN + src) * kD);
      _Float16* q = feat + lw * kLdsStride;
#pragma unroll
      for (int i = 0; i < 8; ++i) {
        const float4 v = p[i];
        q[i * 4 + 0] = (_Float16)v.x; q[i * 4 + 1] = (_Float16)v.y;
        q[i * 4 + 2] = (_Float16)v.z; q[i * 4 + 3] = (_Float16)v.w;
      }
    } else {
      const int dst = to_ind[e];
      s_to[wave][lw] = dst;
      const float4* p = (const float4*)(h_to + ((size_t)b * kNumCN + dst) * kD);
      _Float16* q = feat + lw * kLdsStride + 32;
#pragma unroll
      for (int i = 0; i < 8; ++i) {
        const float4 v = p[i];
        q[i * 4 + 0] = (_Float16)v.x; q[i * 4 + 1] = (_Float16)v.y;
        q[i * 4 + 2] = (_Float16)v.z; q[i * 4 + 3] = (_Float16)v.w;
      }
    }
    __builtin_amdgcn_wave_barrier();  // intra-wave: DS pipe is in-order, just pin scheduling

    // Layer 1: hid[16x40] = relu(feat[16x64] @ Wm1[64x40])
    const v16h a0 = load_afrag(feat, 0, lane);
    const v16h a1 = load_afrag(feat, 1, lane);
    v8f acc0 = {}, acc1 = {}, acc2 = {};
    acc0 = WMMA_F16(a0, b1[0][0], acc0); acc0 = WMMA_F16(a1, b1[1][0], acc0);
    acc1 = WMMA_F16(a0, b1[0][1], acc1); acc1 = WMMA_F16(a1, b1[1][1], acc1);
    acc2 = WMMA_F16(a0, b1[0][2], acc2); acc2 = WMMA_F16(a1, b1[1][2], acc2);

    // ReLU + write back in C-layout (lane = column lw, rows hi*8+i).
    // Unconditional for all 48 cols: cols 40..47 are annihilated by the
    // zero-padded k>=40 rows of b2, so no guard (no exec-mask branch) needed.
#pragma unroll
    for (int i = 0; i < 8; ++i) {
      const int r = (hi * 8 + i) * kLdsStride;
      hid[r + lw]      = (_Float16)fmaxf(acc0[i], 0.0f);
      hid[r + 16 + lw] = (_Float16)fmaxf(acc1[i], 0.0f);
      hid[r + 32 + lw] = (_Float16)fmaxf(acc2[i], 0.0f);
    }
    __builtin_amdgcn_wave_barrier();

    // Layer 2: msg[16x20] = hid[16x40(pad64)] @ Wm2
    const v16h h0 = load_afrag(hid, 0, lane);
    const v16h h1 = load_afrag(hid, 1, lane);
    v8f mg0 = {}, mg1 = {};
    mg0 = WMMA_F16(h0, b2[0][0], mg0); mg0 = WMMA_F16(h1, b2[1][0], mg0);
    mg1 = WMMA_F16(h0, b2[0][1], mg1); mg1 = WMMA_F16(h1, b2[1][1], mg1);

    // Scatter-add messages onto receiving check nodes (L2 atomic units).
    float* mb = m + (size_t)b * kNumCN * kMSG;
#pragma unroll
    for (int i = 0; i < 8; ++i) {
      const int row = hi * 8 + i;
      const int dst = s_to[wave][row];
      atomicAdd(&mb[(size_t)dst * kMSG + lw], mg0[i]);
      if (lw < kMSG - 16) atomicAdd(&mb[(size_t)dst * kMSG + 16 + lw], mg1[i]);
    }
  }
}

// ---------------------------------------------------------------------------
// Node kernel: one wave processes a tile of 16 check nodes of one batch element.
// emb_in = [m(20) | h_to(32) | logit(1)] -> relu(@We1[53x40]) -> @We2[40x32]
// ---------------------------------------------------------------------------
__global__ __launch_bounds__(256) void bp_node_kernel(
    const float* __restrict__ m,      // [B, NUM_CN, MSG]
    const float* __restrict__ h_to,   // [B, NUM_CN, D]
    const float* __restrict__ logit,  // [B, NUM_CN]
    const float* __restrict__ We1,    // [53, 40]
    const float* __restrict__ We2,    // [40, 32]
    float*       __restrict__ out)    // [B, NUM_CN, D]
{
  __shared__ __align__(16) _Float16 s_emb[kWaves][16 * kLdsStride];
  __shared__ __align__(16) _Float16 s_hid[kWaves][16 * kLdsStride];

  const int wave = threadIdx.x >> 5;
  const int lane = threadIdx.x & 31;
  const int lw   = lane & 15;
  const int hi   = lane >> 4;

  _Float16* emb = s_emb[wave];
  _Float16* hid = s_hid[wave];

  for (int i = lane; i < 16 * kLdsStride; i += 32) {
    emb[i] = (_Float16)0.0f;
    hid[i] = (_Float16)0.0f;
  }

  constexpr int kKin = kMSG + kD + 1;  // 53
  v16h b1[2][3];  // We1: K=53 pad 64 (zeros), N=40
#pragma unroll
  for (int kb = 0; kb < 2; ++kb)
#pragma unroll
    for (int nt = 0; nt < 3; ++nt)
      b1[kb][nt] = load_bfrag(We1, kHID, kKin, kHID, kb, nt, lane);
  v16h b2[2][2];  // We2: K=40 pad 64 (zeros), N=32
#pragma unroll
  for (int kb = 0; kb < 2; ++kb)
#pragma unroll
    for (int nt = 0; nt < 2; ++nt)
      b2[kb][nt] = load_bfrag(We2, kD, kHID, kD, kb, nt, lane);

  constexpr int kTilesPerB = kNumCN / 16;      // 2048
  constexpr int kTiles     = kB * kTilesPerB;  // 8192
  const int wid = blockIdx.x * kWaves + wave;
  const int nw  = gridDim.x * kWaves;

  for (int t = wid; t < kTiles; t += nw) {
    const int b    = t / kTilesPerB;
    const int n0   = (t % kTilesPerB) * 16;
    const int node = n0 + lw;

    if (hi == 0) {
      // h_to row -> cols 20..51
      const float4* p = (const float4*)(h_to + ((size_t)b * kNumCN + node) * kD);
      _Float16* q = emb + lw * kLdsStride + kMSG;
#pragma unroll
      for (int i = 0; i < 8; ++i) {
        const float4 v = p[i];
        q[i * 4 + 0] = (_Float16)v.x; q[i * 4 + 1] = (_Float16)v.y;
        q[i * 4 + 2] = (_Float16)v.z; q[i * 4 + 3] = (_Float16)v.w;
      }
    } else {
      // m row -> cols 0..19 (rows are 80B = 16B-aligned), logit -> col 52
      const float4* p = (const float4*)(m + ((size_t)b * kNumCN + node) * kMSG);
      _Float16* q = emb + lw * kLdsStride;
#pragma unroll
      for (int i = 0; i < 5; ++i) {
        const float4 v = p[i];
        q[i * 4 + 0] = (_Float16)v.x; q[i * 4 + 1] = (_Float16)v.y;
        q[i * 4 + 2] = (_Float16)v.z; q[i * 4 + 3] = (_Float16)v.w;
      }
      q[kMSG + kD] = (_Float16)logit[(size_t)b * kNumCN + node];
    }
    __builtin_amdgcn_wave_barrier();

    const v16h a0 = load_afrag(emb, 0, lane);
    const v16h a1 = load_afrag(emb, 1, lane);
    v8f acc0 = {}, acc1 = {}, acc2 = {};
    acc0 = WMMA_F16(a0, b1[0][0], acc0); acc0 = WMMA_F16(a1, b1[1][0], acc0);
    acc1 = WMMA_F16(a0, b1[0][1], acc1); acc1 = WMMA_F16(a1, b1[1][1], acc1);
    acc2 = WMMA_F16(a0, b1[0][2], acc2); acc2 = WMMA_F16(a1, b1[1][2], acc2);

    // Unconditional ReLU write-back (cols 40..47 killed by zero-padded b2).
#pragma unroll
    for (int i = 0; i < 8; ++i) {
      const int r = (hi * 8 + i) * kLdsStride;
      hid[r + lw]      = (_Float16)fmaxf(acc0[i], 0.0f);
      hid[r + 16 + lw] = (_Float16)fmaxf(acc1[i], 0.0f);
      hid[r + 32 + lw] = (_Float16)fmaxf(acc2[i], 0.0f);
    }
    __builtin_amdgcn_wave_barrier();

    const v16h h0 = load_afrag(hid, 0, lane);
    const v16h h1 = load_afrag(hid, 1, lane);
    v8f o0 = {}, o1 = {};
    o0 = WMMA_F16(h0, b2[0][0], o0); o0 = WMMA_F16(h1, b2[1][0], o0);
    o1 = WMMA_F16(h0, b2[0][1], o1); o1 = WMMA_F16(h1, b2[1][1], o1);

    float* ob = out + ((size_t)b * kNumCN + n0) * kD;
#pragma unroll
    for (int i = 0; i < 8; ++i) {
      const int row = hi * 8 + i;
      ob[row * kD + lw]      = o0[i];
      ob[row * kD + 16 + lw] = o1[i];
    }
  }
}

// ---------------------------------------------------------------------------
extern "C" void kernel_launch(void* const* d_in, const int* in_sizes, int n_in,
                              void* d_out, int out_size, void* d_ws, size_t ws_size,
                              hipStream_t stream) {
  (void)in_sizes; (void)n_in; (void)out_size; (void)ws_size;

  const float* h_from     = (const float*)d_in[0];
  const float* h_to_x     = (const float*)d_in[1];
  const float* h_to_z     = (const float*)d_in[2];
  const float* hx_logit   = (const float*)d_in[3];
  const float* hz_logit   = (const float*)d_in[4];
  const int*   from_ind_x = (const int*)d_in[5];
  const int*   to_ind_x   = (const int*)d_in[6];
  const int*   from_ind_z = (const int*)d_in[7];
  const int*   to_ind_z   = (const int*)d_in[8];
  const float* Wm1_x      = (const float*)d_in[9];
  const float* Wm2_x      = (const float*)d_in[10];
  const float* Wm1_z      = (const float*)d_in[11];
  const float* Wm2_z      = (const float*)d_in[12];
  const float* We1_x      = (const float*)d_in[13];
  const float* We2_x      = (const float*)d_in[14];
  const float* We1_z      = (const float*)d_in[15];
  const float* We2_z      = (const float*)d_in[16];

  float* out  = (float*)d_out;
  float* m_x  = (float*)d_ws;                              // [B,NUM_CN,MSG]
  float* m_z  = m_x + (size_t)kB * kNumCN * kMSG;          // [B,NUM_CN,MSG]

  const int mElems = 2 * kB * kNumCN * kMSG;               // 5,242,880 (21 MB)
  bp_zero_kernel<<<(mElems + 255) / 256, 256, 0, stream>>>(m_x, mElems);

  bp_edge_kernel<<<1024, 256, 0, stream>>>(h_from, h_to_x, from_ind_x, to_ind_x,
                                           Wm1_x, Wm2_x, m_x);
  bp_edge_kernel<<<1024, 256, 0, stream>>>(h_from, h_to_z, from_ind_z, to_ind_z,
                                           Wm1_z, Wm2_z, m_z);

  bp_node_kernel<<<256, 256, 0, stream>>>(m_x, h_to_x, hx_logit, We1_x, We2_x,
                                          out);
  bp_node_kernel<<<256, 256, 0, stream>>>(m_z, h_to_z, hz_logit, We1_z, We2_z,
                                          out + (size_t)kB * kNumCN * kD);
}
